// StrokeGAT_15925738733956
// MI455X (gfx1250) — matrix operand order repaired and checked by reading, not compiled, not run on hardware
//
#include <hip/hip_runtime.h>

// ---------------------------------------------------------------------------
// StrokeGAT forward: 2x GAT (H1=4,C1=64 / H2=2,C2=32) + BN+ELU, mean-pool, MLP
// N=100000 nodes, E=900000 edges (incl self loops), G=64 graphs.
// Layer-2 GEMM [N,256]x[256,64] via v_wmma_f32_16x16x32_f16 with a
// fragment-ready pre-permuted B operand (pure b128 loads, no LDS shuffling).
// ---------------------------------------------------------------------------

typedef __attribute__((ext_vector_type(16))) _Float16 v16h;
typedef __attribute__((ext_vector_type(8)))  float    v8f;

#define NEG_SLOPE 0.2f
#define BN_EPS    1e-5f
#define GG        64           // num graphs
#define HC1       256          // H1*C1
#define HC2       64           // H2*C2

__device__ __forceinline__ void atomicMaxF(float* addr, float val) {
  if (val >= 0.0f) atomicMax((int*)addr, __float_as_int(val));
  else             atomicMin((unsigned int*)addr, __float_as_uint(val));
}

// --- prep: W2 f32 -> fragment-ready f16 layout + layer-1 attention consts ---
// W2perm[((w*8+kt)*32 + lane)*16 + j] = W2[k*64 + n]
//   with k = kt*32 + (lane>>4)*16 + j, n = w*16 + (lane&15)
// so in the GEMM each lane's B fragment is 32 contiguous bytes.
__global__ void k_prep(const float* __restrict__ W2, _Float16* __restrict__ W2p,
                       const float* __restrict__ W1, const float* __restrict__ as1,
                       const float* __restrict__ ad1, float* __restrict__ sbuf) {
  int g = blockIdx.x * blockDim.x + threadIdx.x;
  if (g < 256 * 64) {
    int j    = g & 15;
    int lane = (g >> 4) & 31;
    int kt   = (g >> 9) & 7;
    int w    = g >> 12;
    int k = kt * 32 + (lane >> 4) * 16 + j;
    int n = w * 16 + (lane & 15);
    W2p[g] = (_Float16)W2[k * 64 + n];
  }
  if (g < 8) {
    int h = g & 3;
    const float* att = (g < 4) ? as1 : ad1;
    float s = 0.f;
    for (int c = 0; c < 64; ++c) s += W1[h * 64 + c] * att[h * 64 + c];
    sbuf[g] = s;
  }
}

// --- zero a float4 buffer ---------------------------------------------------
__global__ void k_zero4(float4* __restrict__ p, int n4) {
  int i = blockIdx.x * blockDim.x + threadIdx.x;
  if (i < n4) p[i] = float4{0.f, 0.f, 0.f, 0.f};
}

// --- layer-1 per-node: attention scalars + softmax-state init --------------
__global__ void k_node1(const float* __restrict__ x, const float* __restrict__ sbuf,
                        float* __restrict__ a_src, float* __restrict__ a_dst,
                        float* __restrict__ emax, float* __restrict__ denom, int N) {
  int n = blockIdx.x * blockDim.x + threadIdx.x;
  if (n >= N) return;
  float xv = x[n];
#pragma unroll
  for (int h = 0; h < 4; ++h) {
    a_src[n * 4 + h] = xv * sbuf[h];
    a_dst[n * 4 + h] = xv * sbuf[4 + h];
    emax[n * 4 + h]  = -INFINITY;
    denom[n * 4 + h] = 0.f;
  }
}

// --- edge logits + segment max (generic over heads H, power of 2) ----------
template <int H, int LOGH>
__global__ void k_edge_max(const int* __restrict__ src, const int* __restrict__ dst,
                           const float* __restrict__ a_src, const float* __restrict__ a_dst,
                           float* __restrict__ e_store, float* __restrict__ emax, int E) {
  int g = blockIdx.x * blockDim.x + threadIdx.x;
  if (g >= E * H) return;
  int e = g >> LOGH, h = g & (H - 1);
  float v = a_src[src[e] * H + h] + a_dst[dst[e] * H + h];
  v = (v > 0.f) ? v : NEG_SLOPE * v;          // leaky_relu
  e_store[g] = v;
  atomicMaxF(&emax[dst[e] * H + h], v);
}

// --- exp(e - max) in place + segment sum ------------------------------------
template <int H, int LOGH>
__global__ void k_edge_exp(const int* __restrict__ dst, float* __restrict__ e_store,
                           const float* __restrict__ emax, float* __restrict__ denom, int E) {
  int g = blockIdx.x * blockDim.x + threadIdx.x;
  if (g >= E * H) return;
  int e = g >> LOGH, h = g & (H - 1);
  int d = dst[e];
  float v = expf(e_store[g] - emax[d * H + h]);
  e_store[g] = v;
  atomicAdd(&denom[d * H + h], v);
}

// --- layer-1 aggregation: out1[dst,hc] += x[src]*W1[hc]*alpha ----------------
// work item g: e = g>>6, t = g&63 handles channels t*4..t*4+3, head = t>>4
__global__ void __launch_bounds__(256)
k_aggr1(const int* __restrict__ src, const int* __restrict__ dst,
        const float* __restrict__ x, const float* __restrict__ W1,
        const float* __restrict__ ex1, const float* __restrict__ denom1,
        float* __restrict__ out1, long long total) {
  __shared__ float w1s[256];
  if (threadIdx.x < 256) w1s[threadIdx.x] = W1[threadIdx.x];
  __syncthreads();
  long long g = (long long)blockIdx.x * blockDim.x + threadIdx.x;
  if (g >= total) return;
  int t = (int)(g & 63);
  int e = (int)(g >> 6);
  int h = t >> 4;
  int c0 = t * 4;
  int s = src[e], d = dst[e];
  float alpha = ex1[e * 4 + h] / (denom1[d * 4 + h] + 1e-16f);
  float scale = x[s] * alpha;
  float* o = out1 + (long long)d * HC1 + c0;
  atomicAdd(o + 0, scale * w1s[c0 + 0]);
  atomicAdd(o + 1, scale * w1s[c0 + 1]);
  atomicAdd(o + 2, scale * w1s[c0 + 2]);
  atomicAdd(o + 3, scale * w1s[c0 + 3]);
}

// --- post layer 1: +bias, BN, ELU -> f16 feature matrix ---------------------
__global__ void k_post1(const float* __restrict__ out1, const float* __restrict__ b1,
                        const float* __restrict__ g1, const float* __restrict__ be1,
                        const float* __restrict__ m1, const float* __restrict__ v1,
                        _Float16* __restrict__ h1h, int N) {
  int g = blockIdx.x * blockDim.x + threadIdx.x;   // one per (node, 4-chan)
  if (g >= N * 64) return;
  int n = g >> 6;
  int c0 = (g & 63) * 4;
  float4 v = *(const float4*)(out1 + (long long)n * HC1 + c0);
  float4 bb = *(const float4*)(b1 + c0), gg = *(const float4*)(g1 + c0);
  float4 be = *(const float4*)(be1 + c0), mm = *(const float4*)(m1 + c0);
  float4 vv = *(const float4*)(v1 + c0);
  float vals[4] = {v.x + bb.x, v.y + bb.y, v.z + bb.z, v.w + bb.w};
  float gam[4] = {gg.x, gg.y, gg.z, gg.w}, bet[4] = {be.x, be.y, be.z, be.w};
  float mu[4] = {mm.x, mm.y, mm.z, mm.w}, var[4] = {vv.x, vv.y, vv.z, vv.w};
  union { _Float16 h[4]; uint2 u; } o;
#pragma unroll
  for (int i = 0; i < 4; ++i) {
    float t = gam[i] * (vals[i] - mu[i]) * rsqrtf(var[i] + BN_EPS) + bet[i];
    t = (t > 0.f) ? t : (expf(t) - 1.f);           // ELU
    o.h[i] = (_Float16)t;
  }
  *(uint2*)(h1h + (long long)n * HC1 + c0) = o.u;
}

// --- layer-2 GEMM: [N,256](f16) x [256,64](f16) -> [N,64](f32) via WMMA -----
// grid.x = N/32, block = 128 (4 waves). Wave w owns cols [16w,16w+16) and
// hoists all 8 pre-permuted B fragments into registers (2 b128 loads each),
// then runs 2 M-tiles of 16 rows: 16 WMMAs per wave, all-b128 operand loads.
__global__ void __launch_bounds__(128)
k_gemm(const _Float16* __restrict__ A, const _Float16* __restrict__ Bp,
       float* __restrict__ C, int N) {
  int wave = threadIdx.x >> 5, lane = threadIdx.x & 31;
  int ml = lane & 15, hi = lane >> 4;
  // hoisted B fragments: contiguous 32B per (wave, kt, lane)
  v16h bfr[8];
  const _Float16* bbase = Bp + (size_t)wave * 4096 + (size_t)lane * 16;
#pragma unroll
  for (int kt = 0; kt < 8; ++kt) {
    union { v16h v; uint4 u[2]; } bf;
    bf.u[0] = *(const uint4*)(bbase + kt * 512);
    bf.u[1] = *(const uint4*)(bbase + kt * 512 + 8);
    bfr[kt] = bf.v;
  }
  int col = wave * 16 + ml;
#pragma unroll
  for (int mt = 0; mt < 2; ++mt) {
    int row = blockIdx.x * 32 + mt * 16 + ml;
    int rowc = (row < N) ? row : (N - 1);          // clamp (N%32==0 normally)
    const _Float16* arow = A + (long long)rowc * HC1;
    v8f acc = {0.f, 0.f, 0.f, 0.f, 0.f, 0.f, 0.f, 0.f};
#pragma unroll
    for (int kt = 0; kt < 8; ++kt) {
      int ka0 = kt * 32 + hi * 16;                 // A: lanes0-15 K0..15, lanes16-31 K16..31
      union { v16h v; uint4 u[2]; } af;
      af.u[0] = *(const uint4*)(arow + ka0);
      af.u[1] = *(const uint4*)(arow + ka0 + 8);
      acc = __builtin_amdgcn_wmma_f32_16x16x32_f16(
          false, af.v, false, bfr[kt], (short)0, acc, false, false);
    }
#pragma unroll
    for (int r = 0; r < 8; ++r) {
      int orow = blockIdx.x * 32 + mt * 16 + hi * 8 + r;  // C/D: VGPR r -> M
      if (orow < N) C[(long long)orow * HC2 + col] = acc[r];
    }
  }
}

// --- layer-2 per-node: attention scalars, init accumulators, zero pooled ----
__global__ void k_node2(const float* __restrict__ h2, const float* __restrict__ as2,
                        const float* __restrict__ ad2, float* __restrict__ a_src,
                        float* __restrict__ a_dst, float* __restrict__ emax,
                        float* __restrict__ denom, float* __restrict__ out2,
                        float* __restrict__ pooled, int N) {
  int g = blockIdx.x * blockDim.x + threadIdx.x;
  if (g < GG * 64 + GG) pooled[g] = 0.f;           // zero pooled sums + counts
  if (g >= N) return;
  const float* row = h2 + (long long)g * HC2;
  float s0 = 0.f, s1 = 0.f, d0 = 0.f, d1 = 0.f;
#pragma unroll
  for (int c = 0; c < 32; ++c) { s0 += row[c] * as2[c];      d0 += row[c] * ad2[c]; }
#pragma unroll
  for (int c = 0; c < 32; ++c) { s1 += row[32 + c] * as2[32 + c]; d1 += row[32 + c] * ad2[32 + c]; }
  a_src[g * 2] = s0; a_src[g * 2 + 1] = s1;
  a_dst[g * 2] = d0; a_dst[g * 2 + 1] = d1;
  emax[g * 2] = -INFINITY; emax[g * 2 + 1] = -INFINITY;
  denom[g * 2] = 0.f; denom[g * 2 + 1] = 0.f;
  float4* o = (float4*)(out2 + (long long)g * HC2);
#pragma unroll
  for (int i = 0; i < 16; ++i) o[i] = float4{0.f, 0.f, 0.f, 0.f};
}

// --- layer-2 aggregation: out2[dst,c] += h2[src,c]*alpha ---------------------
// work item g: e = g>>4, t = g&15 handles channels t*4..t*4+3, head = t>>3
__global__ void __launch_bounds__(256)
k_aggr2(const int* __restrict__ src, const int* __restrict__ dst,
        const float* __restrict__ h2, const float* __restrict__ ex2,
        const float* __restrict__ denom2, float* __restrict__ out2, long long total) {
  long long g = (long long)blockIdx.x * blockDim.x + threadIdx.x;
  if (g >= total) return;
  int t = (int)(g & 15);
  int e = (int)(g >> 4);
  int h = t >> 3;
  int c0 = t * 4;
  int s = src[e], d = dst[e];
  float alpha = ex2[e * 2 + h] / (denom2[d * 2 + h] + 1e-16f);
  float4 hv = *(const float4*)(h2 + (long long)s * HC2 + c0);
  float* o = out2 + (long long)d * HC2 + c0;
  atomicAdd(o + 0, hv.x * alpha);
  atomicAdd(o + 1, hv.y * alpha);
  atomicAdd(o + 2, hv.z * alpha);
  atomicAdd(o + 3, hv.w * alpha);
}

// --- post layer 2 + LDS-hierarchical mean-pool accumulation -----------------
__global__ void __launch_bounds__(256)
k_pool(const float* __restrict__ out2, const int* __restrict__ batch,
       const float* __restrict__ b2, const float* __restrict__ g2,
       const float* __restrict__ be2, const float* __restrict__ m2,
       const float* __restrict__ v2, float* __restrict__ pooled, int N) {
  __shared__ float lp[GG * 64 + GG];
  for (int i = threadIdx.x; i < GG * 64 + GG; i += 256) lp[i] = 0.f;
  __syncthreads();
  int n = blockIdx.x * 256 + threadIdx.x;
  if (n < N) {
    int b = batch[n];
    const float* row = out2 + (long long)n * HC2;
#pragma unroll 4
    for (int c = 0; c < 64; ++c) {
      float val = row[c] + b2[c];
      val = g2[c] * (val - m2[c]) * rsqrtf(v2[c] + BN_EPS) + be2[c];
      val = (val > 0.f) ? val : (expf(val) - 1.f);  // ELU
      atomicAdd(&lp[b * 64 + c], val);
    }
    atomicAdd(&lp[GG * 64 + b], 1.f);
  }
  __syncthreads();
  for (int i = threadIdx.x; i < GG * 64 + GG; i += 256) {
    float v = lp[i];
    if (v != 0.f) atomicAdd(&pooled[i], v);
  }
}

// --- classifier head: mean, relu(pooled@Wc1+bc1)@Wc2+bc2; write outputs -----
__global__ void k_head(const float* __restrict__ pooled, const float* __restrict__ Wc1,
                       const float* __restrict__ bc1, const float* __restrict__ Wc2,
                       const float* __restrict__ bc2, float* __restrict__ out) {
  __shared__ float pl[GG * 64];
  int t = threadIdx.x;                              // 64 threads, one per graph
  float cnt = fmaxf(pooled[GG * 64 + t], 1.f);
  for (int c = 0; c < 64; ++c) pl[t * 64 + c] = pooled[t * 64 + c] / cnt;
  float logit = bc2[0];
  for (int j = 0; j < 32; ++j) {
    float z = bc1[j];
    for (int c = 0; c < 64; ++c) z += pl[t * 64 + c] * Wc1[c * 32 + j];
    z = fmaxf(z, 0.f);
    logit += z * Wc2[j];
  }
  out[t] = logit;                                   // logits [G]
  for (int c = 0; c < 64; ++c) out[GG + t * 64 + c] = pl[t * 64 + c];  // pooled
}

// ---------------------------------------------------------------------------
extern "C" void kernel_launch(void* const* d_in, const int* in_sizes, int n_in,
                              void* d_out, int out_size, void* d_ws, size_t ws_size,
                              hipStream_t stream) {
  const float* x   = (const float*)d_in[0];
  const int*   ei  = (const int*)d_in[1];
  const int*   bat = (const int*)d_in[2];
  const float* W1  = (const float*)d_in[4];
  const float* as1 = (const float*)d_in[5];
  const float* ad1 = (const float*)d_in[6];
  const float* b1  = (const float*)d_in[7];
  const float* g1  = (const float*)d_in[8];
  const float* be1 = (const float*)d_in[9];
  const float* m1  = (const float*)d_in[10];
  const float* v1  = (const float*)d_in[11];
  const float* W2  = (const float*)d_in[12];
  const float* as2 = (const float*)d_in[13];
  const float* ad2 = (const float*)d_in[14];
  const float* b2  = (const float*)d_in[15];
  const float* g2  = (const float*)d_in[16];
  const float* be2 = (const float*)d_in[17];
  const float* m2  = (const float*)d_in[18];
  const float* v2  = (const float*)d_in[19];
  const float* Wc1 = (const float*)d_in[20];
  const float* bc1 = (const float*)d_in[21];
  const float* Wc2 = (const float*)d_in[22];
  const float* bc2 = (const float*)d_in[23];

  const int N = in_sizes[0];          // x is [N,1]
  const int E = in_sizes[1] / 2;      // edge_index [2,E]
  const int* src = ei;
  const int* dst = ei + E;

  // ---- workspace carve (aliasing layer-2 into dead out1 region) ----
  char* ws = (char*)d_ws;
  size_t off = 0;
  auto carve = [&](size_t bytes) -> char* {
    char* p = ws + off;
    off = (off + bytes + 255) & ~(size_t)255;
    return p;
  };
  float*    sbuf   = (float*)carve(8 * sizeof(float));
  float*    a_src1 = (float*)carve((size_t)N * 4 * sizeof(float));
  float*    a_dst1 = (float*)carve((size_t)N * 4 * sizeof(float));
  float*    emax1  = (float*)carve((size_t)N * 4 * sizeof(float));
  float*    denom1 = (float*)carve((size_t)N * 4 * sizeof(float));
  float*    ex1    = (float*)carve((size_t)E * 4 * sizeof(float));
  float*    out1   = (float*)carve((size_t)N * HC1 * sizeof(float));
  _Float16* h1h    = (_Float16*)carve((size_t)N * HC1 * sizeof(_Float16));
  _Float16* W2p    = (_Float16*)carve((size_t)256 * 64 * sizeof(_Float16));
  float*    pooled = (float*)carve((GG * 64 + GG) * sizeof(float));
  // layer-2 buffers live inside out1 (out1 dead after k_post1 consumes it)
  float* h2     = out1;                               // N*64
  float* out2   = out1 + (size_t)N * 64;              // N*64
  float* ex2    = out1 + (size_t)N * 128;             // E*2
  float* a_src2 = ex2 + (size_t)E * 2;                // N*2
  float* a_dst2 = a_src2 + (size_t)N * 2;
  float* emax2  = a_dst2 + (size_t)N * 2;
  float* denom2 = emax2 + (size_t)N * 2;

  const int T = 256;
  // prep + init
  k_prep<<<64, T, 0, stream>>>(W2, W2p, W1, as1, ad1, sbuf);
  k_zero4<<<(N * 64 + T - 1) / T, T, 0, stream>>>((float4*)out1, N * 64);
  k_node1<<<(N + T - 1) / T, T, 0, stream>>>(x, sbuf, a_src1, a_dst1, emax1, denom1, N);
  // layer 1 edge phase
  k_edge_max<4, 2><<<(E * 4 + T - 1) / T, T, 0, stream>>>(src, dst, a_src1, a_dst1, ex1, emax1, E);
  k_edge_exp<4, 2><<<(E * 4 + T - 1) / T, T, 0, stream>>>(dst, ex1, emax1, denom1, E);
  {
    long long total = (long long)E * 64;
    k_aggr1<<<(unsigned)((total + T - 1) / T), T, 0, stream>>>(src, dst, x, W1, ex1, denom1, out1, total);
  }
  k_post1<<<(N * 64 + T - 1) / T, T, 0, stream>>>(out1, b1, g1, be1, m1, v1, h1h, N);
  // layer 2 GEMM (WMMA)
  k_gemm<<<(N + 31) / 32, 128, 0, stream>>>(h1h, W2p, h2, N);
  // layer 2 edge phase
  k_node2<<<(N + T - 1) / T, T, 0, stream>>>(h2, as2, ad2, a_src2, a_dst2, emax2, denom2, out2, pooled, N);
  k_edge_max<2, 1><<<(E * 2 + T - 1) / T, T, 0, stream>>>(src, dst, a_src2, a_dst2, ex2, emax2, E);
  k_edge_exp<2, 1><<<(E * 2 + T - 1) / T, T, 0, stream>>>(dst, ex2, emax2, denom2, E);
  {
    long long total = (long long)E * 16;
    k_aggr2<<<(unsigned)((total + T - 1) / T), T, 0, stream>>>(src, dst, h2, ex2, denom2, out2, total);
  }
  // pooling + head
  k_pool<<<(N + T - 1) / T, T, 0, stream>>>(out2, bat, b2, g2, be2, m2, v2, pooled, N);
  k_head<<<1, 64, 0, stream>>>(pooled, Wc1, bc1, Wc2, bc2, (float*)d_out);
}
